// GATLayer_65300682768656
// MI455X (gfx1250) — matrix operand order repaired and checked by reading, not compiled, not run on hardware
//
#include <hip/hip_runtime.h>

typedef __attribute__((ext_vector_type(16))) _Float16 v16h;
typedef __attribute__((ext_vector_type(8)))  float    v8f;

typedef unsigned int u32x4 __attribute__((ext_vector_type(4)));
typedef int          i32x8 __attribute__((ext_vector_type(8)));
typedef int          i32x4 __attribute__((ext_vector_type(4)));

#if __has_builtin(__builtin_amdgcn_tensor_load_to_lds) && __has_builtin(__builtin_amdgcn_s_wait_tensorcnt)
#define USE_TDM 1
#else
#define USE_TDM 0
#endif

constexpr int NB  = 4;
constexpr int NN  = 512;
constexpr int FIN = 128;
constexpr int NH  = 4;
constexpr int ND  = 32;
constexpr int HID = 128;

#define LN_EPS     1e-5f
#define NEG_SLOPE  0.2f
#define ATT_SCALE  0.17677669529663687f   /* 1/sqrt(32) */

// ---- dynamic-LDS layout for attn_kernel (padded strides to dodge bank conflicts)
constexpr int UT_LD  = 33;   // f32 U_tgt rows  (TDM pad: 1 dword per 32 dwords)
constexpr int UTH_LD = 34;   // f16 U_tgt rows (WMMA B source)
constexpr int PM_LD  = 528;  // f16 alpha rows (WMMA A source)
constexpr int US_LD  = 33;   // f32 U_src tile rows
constexpr size_t OFF_UT   = 0;
constexpr size_t OFF_UTH  = OFF_UT  + (size_t)NN * UT_LD  * sizeof(float);     // 67584
constexpr size_t OFF_PM   = OFF_UTH + (size_t)NN * UTH_LD * sizeof(_Float16);  // 102400
constexpr size_t OFF_US   = OFF_PM  + (size_t)32 * PM_LD  * sizeof(_Float16);  // 136192
constexpr size_t OFF_AV   = OFF_US  + (size_t)32 * US_LD  * sizeof(float);     // 140416
constexpr size_t OFF_RED  = OFF_AV  + 32 * sizeof(float);                      // 140544
constexpr size_t SMEM_ATTN = OFF_RED + 256 * sizeof(float);                    // 141568 B

// =====================================================================
// Kernel 1: U = LeakyReLU(LayerNorm_D(X @ W^T)) for W_src and W_tgt.
// =====================================================================
__global__ __launch_bounds__(256) void proj_kernel(
    const float* __restrict__ X,
    const float* __restrict__ Wsrc, const float* __restrict__ gsrc, const float* __restrict__ bsrc,
    const float* __restrict__ Wtgt, const float* __restrict__ gtgt, const float* __restrict__ btgt,
    float* __restrict__ Usrc, float* __restrict__ Utgt)
{
  __shared__ _Float16 Xh[32][136];    // A-matrix staging (f16)
  __shared__ float    Uacc[32][257];  // cols 0..127 = src, 128..255 = tgt

  const int tid  = threadIdx.x;
  const int row0 = blockIdx.x * 32;

  for (int q = tid; q < 32 * 32; q += 256) {
    int r = q >> 5, c4 = q & 31;
    float4 xv = ((const float4*)(X + (size_t)(row0 + r) * FIN))[c4];
    Xh[r][c4*4+0] = (_Float16)xv.x;
    Xh[r][c4*4+1] = (_Float16)xv.y;
    Xh[r][c4*4+2] = (_Float16)xv.z;
    Xh[r][c4*4+3] = (_Float16)xv.w;
  }
  __syncthreads();

  const int wv  = tid >> 5, lane = tid & 31;
  const int l16 = lane & 15;
  const int ka  = (lane & 16) ? 16 : 0;  // A-frag K base (contiguous k = 16h + s, matches B)
  const int kbB = (lane & 16) ? 16 : 0;  // B-frag K base (16-bit B 32x16)
  const int mc  = (lane & 16) ? 8  : 0;  // C-frag M offset

  for (int tt = 0; tt < 4; ++tt) {
    int T = wv + tt * 8;
    int rowHalf = T & 1;
    int colTile = T >> 1;
    const float* W    = (colTile < 8) ? Wsrc : Wtgt;
    const float* wrow = W + (size_t)((colTile & 7) * 16 + l16) * FIN;  // B[k][n]=W[n][k]
    const int mrow = rowHalf * 16 + l16;

    v8f acc = {};
    for (int k0 = 0; k0 < FIN; k0 += 32) {
      v16h a, bb;
      #pragma unroll
      for (int v = 0; v < 8; ++v) {
        int k = ka + 2 * v;
        a[2*v]   = Xh[mrow][k0 + k];
        a[2*v+1] = Xh[mrow][k0 + k + 1];
      }
      const float4* wp = (const float4*)(wrow + k0 + kbB);
      float4 w0 = wp[0], w1 = wp[1], w2 = wp[2], w3 = wp[3];
      bb[0]=(_Float16)w0.x; bb[1]=(_Float16)w0.y; bb[2]=(_Float16)w0.z; bb[3]=(_Float16)w0.w;
      bb[4]=(_Float16)w1.x; bb[5]=(_Float16)w1.y; bb[6]=(_Float16)w1.z; bb[7]=(_Float16)w1.w;
      bb[8]=(_Float16)w2.x; bb[9]=(_Float16)w2.y; bb[10]=(_Float16)w2.z; bb[11]=(_Float16)w2.w;
      bb[12]=(_Float16)w3.x; bb[13]=(_Float16)w3.y; bb[14]=(_Float16)w3.z; bb[15]=(_Float16)w3.w;
      acc = __builtin_amdgcn_wmma_f32_16x16x32_f16(false, a, false, bb, (short)0, acc, false, false);
    }
    #pragma unroll
    for (int r = 0; r < 8; ++r)
      Uacc[rowHalf*16 + mc + r][colTile*16 + l16] = acc[r];
  }
  __syncthreads();

  {
    const int row = tid & 31;
    const int g   = tid >> 5;
    const int isSrc = (g < 4);
    const int h   = g & 3;
    const int base = (isSrc ? 0 : 128) + h * 32;

    float mu = 0.f;
    #pragma unroll
    for (int d = 0; d < 32; ++d) mu += Uacc[row][base + d];
    mu *= (1.f / 32.f);
    float var = 0.f;
    #pragma unroll
    for (int d = 0; d < 32; ++d) { float dx = Uacc[row][base + d] - mu; var += dx * dx; }
    var *= (1.f / 32.f);
    const float rstd = rsqrtf(var + LN_EPS);

    const int gr = row0 + row;
    const int b = gr >> 9, n = gr & (NN - 1);
    const float* gam = isSrc ? gsrc : gtgt;
    const float* bet = isSrc ? bsrc : btgt;
    float* Uo = isSrc ? Usrc : Utgt;
    const size_t ob = (((size_t)b * NH + h) * NN + n) * ND;
    #pragma unroll
    for (int d = 0; d < 32; ++d) {
      float y = (Uacc[row][base + d] - mu) * rstd * gam[h*32 + d] + bet[h*32 + d];
      y = (y > 0.f) ? y : NEG_SLOPE * y;
      Uo[ob + d] = y;
    }
  }
}

// =====================================================================
// Kernel 2: Res = LeakyReLU(X @ W_res^T + b_res)
// =====================================================================
__global__ __launch_bounds__(256) void res_kernel(
    const float* __restrict__ X, const float* __restrict__ Wres,
    const float* __restrict__ bres, float* __restrict__ Res)
{
  __shared__ _Float16 Xh[32][136];
  const int tid  = threadIdx.x;
  const int row0 = blockIdx.x * 32;

  for (int q = tid; q < 32 * 32; q += 256) {
    int r = q >> 5, c4 = q & 31;
    float4 xv = ((const float4*)(X + (size_t)(row0 + r) * FIN))[c4];
    Xh[r][c4*4+0] = (_Float16)xv.x;
    Xh[r][c4*4+1] = (_Float16)xv.y;
    Xh[r][c4*4+2] = (_Float16)xv.z;
    Xh[r][c4*4+3] = (_Float16)xv.w;
  }
  __syncthreads();

  const int wv  = tid >> 5, lane = tid & 31, l16 = lane & 15;
  const int ka  = (lane & 16) ? 16 : 0;
  const int kbB = (lane & 16) ? 16 : 0;
  const int mc  = (lane & 16) ? 8  : 0;

  for (int tt = 0; tt < 2; ++tt) {
    int T = wv + tt * 8;
    int rowHalf = T & 1;
    int colTile = T >> 1;
    const float* wrow = Wres + (size_t)(colTile * 16 + l16) * FIN;
    const int mrow = rowHalf * 16 + l16;

    v8f acc = {};
    for (int k0 = 0; k0 < FIN; k0 += 32) {
      v16h a, bb;
      #pragma unroll
      for (int v = 0; v < 8; ++v) {
        int k = ka + 2 * v;
        a[2*v]   = Xh[mrow][k0 + k];
        a[2*v+1] = Xh[mrow][k0 + k + 1];
      }
      const float4* wp = (const float4*)(wrow + k0 + kbB);
      float4 w0 = wp[0], w1 = wp[1], w2 = wp[2], w3 = wp[3];
      bb[0]=(_Float16)w0.x; bb[1]=(_Float16)w0.y; bb[2]=(_Float16)w0.z; bb[3]=(_Float16)w0.w;
      bb[4]=(_Float16)w1.x; bb[5]=(_Float16)w1.y; bb[6]=(_Float16)w1.z; bb[7]=(_Float16)w1.w;
      bb[8]=(_Float16)w2.x; bb[9]=(_Float16)w2.y; bb[10]=(_Float16)w2.z; bb[11]=(_Float16)w2.w;
      bb[12]=(_Float16)w3.x; bb[13]=(_Float16)w3.y; bb[14]=(_Float16)w3.z; bb[15]=(_Float16)w3.w;
      acc = __builtin_amdgcn_wmma_f32_16x16x32_f16(false, a, false, bb, (short)0, acc, false, false);
    }
    const int col = colTile * 16 + l16;
    const float bv = bres[col];
    #pragma unroll
    for (int r = 0; r < 8; ++r) {
      float y = acc[r] + bv;
      y = (y > 0.f) ? y : NEG_SLOPE * y;
      Res[(size_t)(row0 + rowHalf*16 + mc + r) * HID + col] = y;
    }
  }
}

// =====================================================================
// Kernel 3: fused attention per (b, h, 32-row i-tile).
//   TDM async-loads the 64KB U_tgt block into LDS (hardware row padding),
//   VALU e-loop -> in-LDS softmax (f16) -> WMMA agg -> +Res, ReLU.
// =====================================================================
__global__ __launch_bounds__(128) void attn_kernel(
    const float* __restrict__ Usrc, const float* __restrict__ Utgt,
    const float* __restrict__ avec, const float* __restrict__ Res,
    float* __restrict__ Out)
{
  extern __shared__ char smem[];
  float*    Ut  = (float*)   (smem + OFF_UT);    // [512][33] f32
  _Float16* Uth = (_Float16*)(smem + OFF_UTH);   // [512][34] f16 (WMMA B)
  _Float16* Pm  = (_Float16*)(smem + OFF_PM);    // [32][528] f16 (e -> p -> alpha; WMMA A)
  float*    Us  = (float*)   (smem + OFF_US);    // [32][33]  f32
  float*    av  = (float*)   (smem + OFF_AV);    // [32]
  float*    red = (float*)   (smem + OFF_RED);   // [0..127]=rowmax, [128..255]=rowsum

  const int tid = threadIdx.x;
  const int ib = blockIdx.x, h = blockIdx.y, b = blockIdx.z;
  const int i0 = ib * 32;
  const float* UtG = Utgt + ((size_t)b * NH + h) * NN * ND;
  const float* UsG = Usrc + (((size_t)b * NH + h) * NN + i0) * ND;

  // Warm the residual tile for the epilogue (global_prefetch_b8).
  __builtin_prefetch(&Res[((size_t)b * NN + i0 + (tid & 31)) * HID + h * ND], 0, 1);

#if USE_TDM
  // ---- Tensor Data Mover: DMA U_tgt[b,h] (512x32 f32, contiguous 64KB) into
  // LDS with hardware padding of 1 dword per 32 dwords -> stride UT_LD=33.
  if (tid < 32) {   // wave 0 issues the descriptor; TDM ignores EXEC
    const unsigned ldsUt = (unsigned)(size_t)(void*)Ut;          // flat addr truncates to LDS offset
    const unsigned long long ga = (unsigned long long)(size_t)UtG;
    u32x4 g0;
    g0[0] = 1u;                                            // count=1 valid descriptor
    g0[1] = ldsUt;                                         // lds_addr
    g0[2] = (unsigned)(ga & 0xFFFFFFFFu);                  // global_addr[31:0]
    g0[3] = (unsigned)((ga >> 32) & 0x1FFFFFFu) | (2u << 30); // global_addr[56:32] | type=2
    i32x8 g1;
    g1[0] = (int)((2u << 16) | (1u << 20) | (4u << 22));   // data_size=4B, pad_enable, pad_interval=32dw, pad_amount=1dw
    g1[1] = (int)(16384u << 16);                           // tensor_dim0[15:0]=16384
    g1[2] = (int)(1u << 16);                               // tensor_dim0 hi=0, tensor_dim1=1
    g1[3] = (int)(16384u << 16);                           // tensor_dim1 hi=0, tile_dim0=16384
    g1[4] = 1;                                             // tile_dim1=1, tile_dim2=0
    g1[5] = 16384;                                         // tensor_dim0_stride lo
    g1[6] = (int)(16384u << 16);                           // stride0 hi=0, tensor_dim1_stride lo=16384
    g1[7] = 0;                                             // stride1 hi
    i32x4 gz = {};
#if defined(__clang_major__) && (__clang_major__ >= 23)
    i32x8 gz8 = {};
    __builtin_amdgcn_tensor_load_to_lds(g0, g1, gz, gz, gz8, 0);
#else
    __builtin_amdgcn_tensor_load_to_lds(g0, g1, gz, gz, 0);
#endif
    __builtin_amdgcn_s_wait_tensorcnt((short)0);           // s_wait_tensorcnt 0 before barrier
  }
#else
  for (int q = tid; q < NN * ND / 4; q += 128) {
    int j = q >> 3, c4 = q & 7;
    float4 v = ((const float4*)(UtG + (size_t)j * ND))[c4];
    float* d0 = &Ut[j * UT_LD + c4 * 4];
    d0[0] = v.x; d0[1] = v.y; d0[2] = v.z; d0[3] = v.w;
  }
#endif

  // U_src tile + scaled attention vector (overlaps the TDM transfer)
  for (int q = tid; q < 32 * ND / 4; q += 128) {
    int r = q >> 3, c4 = q & 7;
    float4 v = ((const float4*)(UsG + (size_t)r * ND))[c4];
    float* d0 = &Us[r * US_LD + c4 * 4];
    d0[0] = v.x; d0[1] = v.y; d0[2] = v.z; d0[3] = v.w;
  }
  if (tid < 32) av[tid] = avec[h * ND + tid] * ATT_SCALE;
  __syncthreads();

  // f16 copy of U_tgt for the WMMA B fragments (LDS -> LDS convert)
  for (int q = tid; q < NN * ND; q += 128) {
    int j = q >> 5, d = q & 31;
    Uth[j * UTH_LD + d] = (_Float16)Ut[j * UT_LD + d];
  }

  // ---- e[i,j] = sum_d lrelu(s_id + t_jd) * a_d ; 4 threads per row i
  const int i = tid >> 2, jl = tid & 3;
  {
    const float* sr = &Us[i * US_LD];
    float m = -1e30f;
    for (int j = jl; j < NN; j += 4) {
      const float* tr = &Ut[j * UT_LD];
      float s = 0.f;
      #pragma unroll
      for (int d = 0; d < ND; ++d) {
        float x = sr[d] + tr[d];
        x = (x > 0.f) ? x : NEG_SLOPE * x;
        s = fmaf(x, av[d], s);
      }
      Pm[i * PM_LD + j] = (_Float16)s;
      m = fmaxf(m, s);
    }
    red[i * 4 + jl] = m;
  }
  __syncthreads();
  {
    const float M = fmaxf(fmaxf(red[i*4+0], red[i*4+1]), fmaxf(red[i*4+2], red[i*4+3]));
    float sum = 0.f;
    for (int j = jl; j < NN; j += 4) {
      float p = __expf((float)Pm[i * PM_LD + j] - M);
      Pm[i * PM_LD + j] = (_Float16)p;
      sum += p;
    }
    red[128 + i * 4 + jl] = sum;
  }
  __syncthreads();
  {
    const float tot = red[128+i*4+0] + red[128+i*4+1] + red[128+i*4+2] + red[128+i*4+3];
    const float inv = 1.f / tot;
    for (int j = jl; j < NN; j += 4)
      Pm[i * PM_LD + j] = (_Float16)((float)Pm[i * PM_LD + j] * inv);
  }
  __syncthreads();

  // ---- agg = alpha[32x512] @ U_tgt[512x32] via WMMA; 4 waves = 4 16x16 tiles
  const int wv = tid >> 5, lane = tid & 31, l16 = lane & 15;
  const int mH = wv >> 1, nH = wv & 1;
  const int ka  = (lane & 16) ? 16 : 0;
  const int kbB = (lane & 16) ? 16 : 0;
  const int mrow = mH * 16 + l16;
  const int ncol = nH * 16 + l16;

  v8f acc = {};
  for (int k0 = 0; k0 < NN; k0 += 32) {
    v16h a, bb;
    #pragma unroll
    for (int v = 0; v < 8; ++v) {
      int k = ka + 2 * v;
      a[2*v]   = Pm[mrow * PM_LD + k0 + k];
      a[2*v+1] = Pm[mrow * PM_LD + k0 + k + 1];
      int kk = k0 + kbB + 2 * v;
      bb[2*v]   = Uth[kk * UTH_LD + ncol];
      bb[2*v+1] = Uth[(kk + 1) * UTH_LD + ncol];
    }
    acc = __builtin_amdgcn_wmma_f32_16x16x32_f16(false, a, false, bb, (short)0, acc, false, false);
  }

  // ---- epilogue: out = relu(agg + Res)
  const int mc = (lane & 16) ? 8 : 0;
  #pragma unroll
  for (int r = 0; r < 8; ++r) {
    const int gi  = i0 + mH * 16 + mc + r;
    const int col = h * ND + ncol;
    const size_t idx = ((size_t)b * NN + gi) * HID + col;
    float y = acc[r] + Res[idx];
    Out[idx] = fmaxf(y, 0.f);
  }
}

// =====================================================================
extern "C" void kernel_launch(void* const* d_in, const int* in_sizes, int n_in,
                              void* d_out, int out_size, void* d_ws, size_t ws_size,
                              hipStream_t stream) {
  (void)in_sizes; (void)n_in; (void)out_size; (void)ws_size;
  const float* X    = (const float*)d_in[0];
  const float* Wsrc = (const float*)d_in[1];
  const float* gsrc = (const float*)d_in[2];
  const float* bsrc = (const float*)d_in[3];
  const float* Wtgt = (const float*)d_in[4];
  const float* gtgt = (const float*)d_in[5];
  const float* btgt = (const float*)d_in[6];
  const float* avec = (const float*)d_in[7];
  const float* Wres = (const float*)d_in[8];
  const float* bres = (const float*)d_in[9];
  float* Out = (float*)d_out;

  char* ws = (char*)d_ws;
  const size_t uBytes = (size_t)NB * NH * NN * ND * sizeof(float);  // 1 MB each
  float* Usrc = (float*)(ws);
  float* Utgt = (float*)(ws + uBytes);
  float* Res  = (float*)(ws + 2 * uBytes);

  proj_kernel<<<dim3(NB * NN / 32), dim3(256), 0, stream>>>(
      X, Wsrc, gsrc, bsrc, Wtgt, gtgt, btgt, Usrc, Utgt);
  res_kernel<<<dim3(NB * NN / 32), dim3(256), 0, stream>>>(X, Wres, bres, Res);
  attn_kernel<<<dim3(NN / 32, NH, NB), dim3(128), SMEM_ATTN, stream>>>(
      Usrc, Utgt, avec, Res, Out);
}
